// PPNet_39573828665537
// MI455X (gfx1250) — matrix operand order repaired and checked by reading, not compiled, not run on hardware
//
#include <hip/hip_runtime.h>
#include <hip/hip_bf16.h>
#include <math.h>

// ---------------------------------------------------------------------------
// Types for CDNA5 WMMA (wave32): f16 A/B fragments, f32 accumulator.
// ---------------------------------------------------------------------------
typedef __attribute__((ext_vector_type(16))) _Float16 v16h;
typedef __attribute__((ext_vector_type(8)))  float    v8f;
typedef __attribute__((ext_vector_type(4)))  unsigned int v4u;
typedef __attribute__((ext_vector_type(8)))  int      v8i;
typedef __attribute__((ext_vector_type(4)))  int      v4i;

__device__ __forceinline__ float sigm(float x) { return 1.0f / (1.0f + expf(-x)); }

// ---------------------------------------------------------------------------
// Weight repack: f32 [Cout,Cin,3,3] -> f16 A-fragment stream.
// Element index: ((tile*ccN + cc)*32 + lane)*16 + j
//   co  = tile*16 + (lane&15)
//   cio = j>>3 ; pos = (j&7) + (lane>>4)*8 ; ci = 2*cc + cio
//   value = (co<Cout && ci<Cin && pos<9) ? W[(co*Cin+ci)*9+pos] : 0
// ---------------------------------------------------------------------------
__global__ void repack_weights_kernel(const float* __restrict__ W, _Float16* __restrict__ dst,
                                      int Cin, int Cout, int ccN, long total)
{
    long i = (long)blockIdx.x * blockDim.x + threadIdx.x;
    if (i >= total) return;
    int  j    = (int)(i & 15);
    int  lane = (int)((i >> 4) & 31);
    long rest = i >> 9;
    int  cc   = (int)(rest % ccN);
    int  tile = (int)(rest / ccN);
    int  co   = tile * 16 + (lane & 15);
    int  cio  = j >> 3;
    int  pos  = (j & 7) + ((lane >> 4) << 3);
    int  ci   = 2 * cc + cio;
    float v = 0.f;
    if (co < Cout && ci < Cin && pos < 9)
        v = W[((long)co * Cin + ci) * 9 + pos];
    dst[i] = (_Float16)v;
}

// ---------------------------------------------------------------------------
// 3x3 SAME conv, implicit GEMM on WMMA.
//   D[co, pix] = sum_k A[co, k] * B[k, pix],   k = ci*16 + pos (pos 0..8 real)
// One wave -> 16 co x 16 pixels. Block = 8 waves = 32 co x 64 pixels.
// B-side halo staged into per-wave LDS slice; interior tiles use the Tensor
// Data Mover (tensor_load_to_lds, TENSORcnt), boundary tiles a manual loop.
// ---------------------------------------------------------------------------
__global__ __launch_bounds__(256) void conv3x3_wmma_kernel(
    const float* __restrict__ in, long inBS,
    const _Float16* __restrict__ Apk,   // repacked weights (A-fragment stream)
    const float* __restrict__ bias,
    float* __restrict__ out,
    int Cin, int Cout, int H, int Wd)
{
    // per-wave private staging slice: 2 input channels x (3 rows x 18 cols) halo
    __shared__ float smem[8][2][3][18];

    const int tid    = threadIdx.x;
    const int wv     = tid >> 5;
    const int lane   = tid & 31;
    const int laneHi = lane >> 4;   // 0: lanes 0-15, 1: lanes 16-31
    const int laneLo = lane & 15;

    const int n      = blockIdx.z;
    const int tile   = blockIdx.y * 2 + (wv >> 2);
    const int coTile = tile * 16;
    const int pix0   = blockIdx.x * 64 + (wv & 3) * 16;  // 16|W so tile is in one row
    const int y      = pix0 / Wd;
    const int x0     = pix0 - y * Wd;

    const float* inN = in + (long)n * inBS;
    const int ccN = (Cin + 1) >> 1;   // 2 input channels per K=32 WMMA step
    const v16h* __restrict__ APK = (const v16h*)Apk + (long)tile * ccN * 32 + lane;

    // ---- TDM descriptor invariants (CDNA5 ISA) ----
    const bool interior = (y >= 1) && (y + 1 < H) && (x0 >= 1) && (x0 + 17 <= Wd);
    const unsigned ldsBase = (unsigned)(wv * (2 * 3 * 18 * 4));
    const unsigned hwp = (unsigned)(H * Wd);
    v8i g1 = { 0x20000,                 // data_size=4B, no multicast
               (int)(18u << 16),        // tensor_dim0 = 18 (lo16)
               (int)(3u << 16),         // tensor_dim0 hi16=0 | tensor_dim1 = 3 (lo16)
               (int)(18u << 16),        // tensor_dim1 hi16=0 | tile_dim0 = 18
               (int)(3u | (2u << 16)),  // tile_dim1 = 3 | tile_dim2 = 2
               (int)(unsigned)Wd,       // tensor_dim0_stride lo32 = W
               (int)((hwp & 0xFFFFu) << 16),   // stride0 hi16=0 | tensor_dim1_stride lo16
               (int)(hwp >> 16) };             // tensor_dim1_stride hi32
    v4i g2 = { 2, 0, 0, 0 };            // tensor_dim2 = 2
    v4i g3 = { 0, 0, 0, 0 };
#if defined(__clang_major__) && (__clang_major__ >= 23)
    v8i g4 = { 0, 0, 0, 0, 0, 0, 0, 0 };
#endif

    v8f acc = {0.f, 0.f, 0.f, 0.f, 0.f, 0.f, 0.f, 0.f};

    for (int cc = 0; cc < ccN; ++cc) {
        if (interior && (2 * cc + 1) < Cin) {
            // ---- async tensor DMA: 2ci x 3row x 18col tile -> this wave's LDS slice
            unsigned long long ga = (unsigned long long)(size_t)
                (inN + ((long)(2 * cc) * H + (y - 1)) * Wd + (x0 - 1));
            v4u g0 = { 1u,                              // count=1, user descriptor
                       ldsBase,                         // lds_addr
                       (unsigned)ga,                    // global_addr lo32
                       (unsigned)(ga >> 32) | (2u << 30) };  // addr[56:32] | type=2
#if defined(__clang_major__) && (__clang_major__ >= 23)
            __builtin_amdgcn_tensor_load_to_lds(g0, g1, g2, g3, g4, 0);
#else
            __builtin_amdgcn_tensor_load_to_lds(g0, g1, g2, g3, 0);
#endif
            __builtin_amdgcn_s_wait_tensorcnt(0);
        } else {
            // ---- manual zero-padded staging (boundary tiles / odd-Cin tail) ----
            for (int idx = lane; idx < 108; idx += 32) {
                int cio = idx / 54;
                int rem = idx - cio * 54;
                int r   = rem / 18;
                int col = rem - r * 18;
                int ci  = 2 * cc + cio;
                int yy  = y + r - 1;
                int xx  = x0 + col - 1;
                float v = 0.f;
                if (ci < Cin && (unsigned)yy < (unsigned)H && (unsigned)xx < (unsigned)Wd)
                    v = inN[((long)ci * H + yy) * Wd + xx];
                smem[wv][cio][r][col] = v;   // per-wave slice; DS in-order per wave
            }
        }

        // ---- A fragment: one 32B vector load per lane from repacked stream ----
        v16h a = APK[(long)cc * 32];
        if (cc + 1 < ccN)
            __builtin_prefetch(&APK[(long)(cc + 1) * 32], 0, 1);

        // ---- B fragment: im2col patch from LDS. lane col = pixel column.
        //      elem j: ci = j>>3, pos = (j&7) + laneHi*8 (matches A table order)
        v16h b;
        {
            const int col = laneLo;
            #pragma unroll
            for (int j = 0; j < 16; ++j) {
                float v = 0.f;
                const int cio = j >> 3;
                const int pos = (j & 7) + laneHi * 8;
                if (pos < 9) {
                    int ky = pos / 3, kx = pos - ky * 3;
                    v = smem[wv][cio][ky][col + kx];
                }
                b[j] = (_Float16)v;
            }
        }
        acc = __builtin_amdgcn_wmma_f32_16x16x32_f16(
            /*neg_a=*/false, a, /*neg_b=*/false, b,
            /*c_mod=*/(short)0, acc, /*reuse_a=*/false, /*reuse_b=*/false);
    }

    // ---- epilogue: C/D layout -> lane col = N, vgpr r = M (+8 for hi lanes)
    const int col = laneLo;
    #pragma unroll
    for (int r = 0; r < 8; ++r) {
        int co = coTile + r + laneHi * 8;
        if (co < Cout)
            out[((long)n * Cout + co) * (long)(H * Wd) + pix0 + col] = acc[r] + bias[co];
    }
}

// ---------------------------------------------------------------------------
// Pointwise kernels
// ---------------------------------------------------------------------------
__global__ void zero_kernel(float* p, long n) {
    long i = (long)blockIdx.x * blockDim.x + threadIdx.x;
    if (i < n) p[i] = 0.f;
}

// copy src [N,Csrc,HW] (batch stride srcBS) into dst [N,Cdst,HW] at channel coff
__global__ void pack_kernel(const float* __restrict__ src, long srcBS,
                            float* __restrict__ dst,
                            int Csrc, int hw, int Cdst, int coff, long total)
{
    long i = (long)blockIdx.x * blockDim.x + threadIdx.x;
    if (i >= total) return;
    long rem = i % ((long)Csrc * hw);
    long n   = i / ((long)Csrc * hw);
    long c   = rem / hw;
    long p   = rem % hw;
    dst[((long)n * Cdst + coff + c) * hw + p] = src[n * srcBS + rem];
}

// ConvLSTM gate math. gates [N,4ch,hw]; h,c [N,ch,hw] updated in place (h==P).
__global__ void lstm_kernel(const float* __restrict__ gates,
                            float* __restrict__ h, float* __restrict__ c,
                            int ch, int hw, long total)
{
    long i = (long)blockIdx.x * blockDim.x + threadIdx.x;
    if (i >= total) return;
    long p  = i % hw;
    long t2 = i / hw;
    long cc = t2 % ch;
    long n  = t2 / ch;
    long gb = (long)n * 4 * ch * hw;
    float gi = gates[gb + cc * hw + p];
    float gf = gates[gb + (ch + cc) * hw + p];
    float go = gates[gb + (2L * ch + cc) * hw + p];
    float gg = gates[gb + (3L * ch + cc) * hw + p];
    float cv = sigm(gf) * c[i] + sigm(gi) * tanhf(gg);
    c[i] = cv;
    h[i] = sigm(go) * tanhf(cv);
}

// bilinear x2 upsample, align per reference: pos = j*(n-1)/(2n-1)
__global__ void upsample2_kernel(const float* __restrict__ in, float* __restrict__ out,
                                 int Hin, int Win, long total)
{
    int Hout = 2 * Hin, Wout = 2 * Win;
    float sy = (float)(Hin - 1) / (float)(2 * Hin - 1);
    float sx = (float)(Win - 1) / (float)(2 * Win - 1);
    long i = (long)blockIdx.x * blockDim.x + threadIdx.x;
    if (i >= total) return;
    int  x  = (int)(i % Wout);
    long t1 = i / Wout;
    int  yy = (int)(t1 % Hout);
    long nc = t1 / Hout;
    float py = yy * sy; int y0 = (int)py; int y1 = y0 + 1; if (y1 > Hin - 1) y1 = Hin - 1;
    float wy = py - y0;
    float px = x * sx;  int xa = (int)px; int xb = xa + 1; if (xb > Win - 1) xb = Win - 1;
    float wx = px - xa;
    const float* p = in + nc * (long)(Hin * Win);
    float v00 = p[y0 * Win + xa], v01 = p[y0 * Win + xb];
    float v10 = p[y1 * Win + xa], v11 = p[y1 * Win + xb];
    float t0 = v00 * (1.f - wy) + v10 * wy;
    float t1v = v01 * (1.f - wy) + v11 * wy;
    out[i] = t0 * (1.f - wx) + t1v * wx;
}

// E = cat(relu(P-A), relu(A-P)) : P [N,C,hw], A batch stride aBS, E [N,2C,hw]
__global__ void error_kernel(const float* __restrict__ P, const float* __restrict__ A,
                             long aBS, float* __restrict__ E, int C, int hw, long total)
{
    long i = (long)blockIdx.x * blockDim.x + threadIdx.x;
    if (i >= total) return;
    long p  = i % hw;
    long t2 = i / hw;
    long c  = t2 % C;
    long n  = t2 / C;
    float d = P[i] - A[n * aBS + c * hw + p];
    E[((long)n * 2 * C + c) * hw + p]     = d > 0.f ? d : 0.f;
    E[((long)n * 2 * C + C + c) * hw + p] = d < 0.f ? -d : 0.f;
}

__global__ void maxpool_kernel(const float* __restrict__ in, float* __restrict__ out,
                               int C, int H, int W, long total)
{
    int Ho = H >> 1, Wo = W >> 1;
    long i = (long)blockIdx.x * blockDim.x + threadIdx.x;
    if (i >= total) return;
    int  x  = (int)(i % Wo);
    long t1 = i / Wo;
    int  yv = (int)(t1 % Ho);
    long nc = t1 / Ho;
    const float* p = in + nc * (long)(H * W) + (long)(2 * yv) * W + 2 * x;
    float a = p[0], b = p[1], c = p[W], d = p[W + 1];
    float m = a > b ? a : b;
    float m2 = c > d ? c : d;
    out[i] = m > m2 ? m : m2;
}

__global__ void relu_copy_kernel(const float* __restrict__ src, float* __restrict__ dst, long n)
{
    long i = (long)blockIdx.x * blockDim.x + threadIdx.x;
    if (i < n) { float v = src[i]; dst[i] = v > 0.f ? v : 0.f; }
}

// sum (P - tgt)^2 ; P batch stride chw, tgt batch stride tgtBS
__global__ void loss_partial_kernel(const float* __restrict__ P, const float* __restrict__ tgt,
                                    long tgtBS, float* __restrict__ partial, long total, int chw)
{
    __shared__ float red[256];
    float s = 0.f;
    for (long i = (long)blockIdx.x * blockDim.x + threadIdx.x; i < total;
         i += (long)gridDim.x * blockDim.x) {
        long n = i / chw, r = i % chw;
        float d = P[n * (long)chw + r] - tgt[n * tgtBS + r];
        s += d * d;
    }
    red[threadIdx.x] = s;
    __syncthreads();
    for (int o = 128; o > 0; o >>= 1) {
        if ((int)threadIdx.x < o) red[threadIdx.x] += red[threadIdx.x + o];
        __syncthreads();
    }
    if (threadIdx.x == 0) partial[blockIdx.x] = red[0];
}

__global__ void loss_finalize_kernel(const float* __restrict__ partial, float scale,
                                     float* __restrict__ loss)
{
    __shared__ float red[256];
    red[threadIdx.x] = partial[threadIdx.x];
    __syncthreads();
    for (int o = 128; o > 0; o >>= 1) {
        if ((int)threadIdx.x < o) red[threadIdx.x] += red[threadIdx.x + o];
        __syncthreads();
    }
    if (threadIdx.x == 0) loss[0] += red[0] * scale;
}

// ---------------------------------------------------------------------------
// Host orchestration
// ---------------------------------------------------------------------------
extern "C" void kernel_launch(void* const* d_in, const int* in_sizes, int n_in,
                              void* d_out, int out_size, void* d_ws, size_t ws_size,
                              hipStream_t stream)
{
    (void)in_sizes; (void)n_in; (void)out_size; (void)ws_size;

    const float* inputs = (const float*)d_in[0];   // [4, 8, 3, 128, 128]
    const float *WOHP_W[4], *WOHP_B[4], *WHP_W[3], *WHP_B[3];
    const float *UPD_W[3], *UPD_B[3], *DWN_W[3], *DWN_B[3];
    for (int l = 0; l < 4; ++l) { WOHP_W[l] = (const float*)d_in[1 + 2 * l];  WOHP_B[l] = (const float*)d_in[2 + 2 * l]; }
    for (int l = 0; l < 3; ++l) { WHP_W[l]  = (const float*)d_in[9 + 2 * l];  WHP_B[l]  = (const float*)d_in[10 + 2 * l]; }
    for (int l = 0; l < 3; ++l) { UPD_W[l]  = (const float*)d_in[15 + 2 * l]; UPD_B[l]  = (const float*)d_in[16 + 2 * l]; }
    for (int l = 0; l < 3; ++l) { DWN_W[l]  = (const float*)d_in[21 + 2 * l]; DWN_B[l]  = (const float*)d_in[22 + 2 * l]; }

    const int CH[4]  = {3, 32, 64, 128};
    const int RES[4] = {128, 64, 32, 16};
    const int HW[4]  = {16384, 4096, 1024, 256};

    // ---- workspace carve-out (f32 elements; every chunk is 32B aligned) ----
    float* ws = (float*)d_ws;
    size_t off = 0;
    auto alloc = [&](size_t nelem) { float* p = ws + off; off += nelem; return p; };
    float *h[4], *c[4];
    for (int l = 0; l < 4; ++l) { h[l] = alloc(4L * CH[l] * HW[l]); c[l] = alloc(4L * CH[l] * HW[l]); }
    float *Acur[4] = {nullptr}, *Anext[4] = {nullptr};
    for (int l = 1; l < 4; ++l) { Acur[l] = alloc(4L * CH[l] * HW[l]); Anext[l] = alloc(4L * CH[l] * HW[l]); }
    float* A0buf   = alloc(4L * 3 * 16384);
    float* gates   = alloc(2097152);   // max 4*4*32*4096
    float* cbuf    = alloc(2490368);   // max 4*38*16384
    float* hp      = alloc(2097152);   // max 4*32*16384
    float* dptmp   = alloc(524288);    // max 4*32*4096
    float* Ebuf    = alloc(1048576);   // max 4*64*4096
    float* pooltmp = alloc(2097152);   // max 4*32*16384
    float* partial = alloc(256);

    auto nblk = [](long n) { return (unsigned)((n + 255) / 256); };
    auto zero = [&](float* p, long n) { zero_kernel<<<nblk(n), 256, 0, stream>>>(p, n); };
    auto pack = [&](const float* src, long srcBS, float* dst, int Csrc, int hw, int Cdst, int coff) {
        long total = 4L * Csrc * hw;
        pack_kernel<<<nblk(total), 256, 0, stream>>>(src, srcBS, dst, Csrc, hw, Cdst, coff, total);
    };
    // repack weights once per call into f16 A-fragment stream (lives in L2)
    auto repack = [&](const float* W, int Cin, int Cout) -> const _Float16* {
        int tiles = (Cout + 15) / 16, ccN = (Cin + 1) / 2;
        long totalH = (long)tiles * ccN * 512;         // halfs, multiple of 512
        _Float16* dst = (_Float16*)alloc((size_t)(totalH / 2));
        repack_weights_kernel<<<nblk(totalH), 256, 0, stream>>>(W, dst, Cin, Cout, ccN, totalH);
        return dst;
    };
    auto conv = [&](const float* in, long inBS, const _Float16* apk, const float* b, float* out,
                    int Cin, int Cout, int R) {
        dim3 grid((unsigned)(R * R / 64), (unsigned)((Cout + 31) / 32), 4);
        conv3x3_wmma_kernel<<<grid, 256, 0, stream>>>(in, inBS, apk, b, out, Cin, Cout, R, R);
    };

    // ---- init ----
    for (int l = 0; l < 4; ++l) { zero(h[l], 4L * CH[l] * HW[l]); zero(c[l], 4L * CH[l] * HW[l]); }
    zero((float*)d_out, 1);

    const _Float16 *WOHP_P[4], *WHP_P[3], *UPD_P[3], *DWN_P[3];
    for (int l = 0; l < 4; ++l) WOHP_P[l] = repack(WOHP_W[l], 2 * CH[l], 4 * CH[l]);
    for (int l = 0; l < 3; ++l) WHP_P[l]  = repack(WHP_W[l], 2 * CH[l] + CH[l + 1], 4 * CH[l]);
    for (int l = 0; l < 3; ++l) UPD_P[l]  = repack(UPD_W[l], 3 * CH[l], CH[l + 1]);
    for (int l = 0; l < 3; ++l) DWN_P[l]  = repack(DWN_W[l], CH[l + 1], CH[l + 1]);

    // ---- timestep loop: t = 0..6 (real_t=6, PredSteps=2) ----
    for (int t = 0; t <= 6; ++t) {
        // ---- first inner loop: layers top-down ----
        for (int l = 3; l >= 0; --l) {
            if (t < l) continue;
            const int chl = CH[l], R = RES[l], hwl = HW[l];
            const float* curA; long curBS;
            if (l == 0) {
                if (t < 6) { curA = inputs + (size_t)t * 49152; curBS = 393216; }
                else       { curA = A0buf;                      curBS = 49152;  }
            } else { curA = Acur[l]; curBS = (long)chl * hwl; }

            const bool wohp = (l == 3) || (t == l);
            if (wohp) {
                const int Cin = 2 * chl;
                pack(curA, curBS, cbuf, chl, hwl, Cin, 0);
                pack(h[l], (long)chl * hwl, cbuf, chl, hwl, Cin, chl);
                conv(cbuf, (long)Cin * hwl, WOHP_P[l], WOHP_B[l], gates, Cin, 4 * chl, R);
            } else {
                const int chh = CH[l + 1], Rh = RES[l + 1], hwh = HW[l + 1];
                // hp = upsample2(conv(P[l+1]))  with P[l+1] == h[l+1] (already stepped)
                conv(h[l + 1], (long)chh * hwh, DWN_P[l], DWN_B[l], dptmp, chh, chh, Rh);
                long uptot = 4L * chh * hwl;
                upsample2_kernel<<<nblk(uptot), 256, 0, stream>>>(dptmp, hp, Rh, Rh, uptot);
                const int Cin = 2 * chl + chh;
                pack(curA, curBS, cbuf, chl, hwl, Cin, 0);
                pack(hp, (long)chh * hwl, cbuf, chh, hwl, Cin, chl);
                pack(h[l], (long)chl * hwl, cbuf, chl, hwl, Cin, chl + chh);
                conv(cbuf, (long)Cin * hwl, WHP_P[l], WHP_B[l], gates, Cin, 4 * chl, R);
            }
            long st = 4L * chl * hwl;
            lstm_kernel<<<nblk(st), 256, 0, stream>>>(gates, h[l], c[l], chl, hwl, st);
            if (l == 0 && t == 5)   // listA[0][6] = relu(P0[5])
                relu_copy_kernel<<<nblk(4L * 49152), 256, 0, stream>>>(h[0], A0buf, 4L * 49152);
        }

        // ---- second inner loop: errors / update_A / maxpool (skipped at t=6: outputs unused) ----
        for (int l = 0; l <= 2; ++l) {
            if (t < l || t > 5) continue;
            const int chl = CH[l], R = RES[l], hwl = HW[l];
            const float* curA; long curBS;
            if (l == 0) { curA = inputs + (size_t)t * 49152; curBS = 393216; }
            else        { curA = Acur[l]; curBS = (long)chl * hwl; }
            const float* nA; long nABS;
            if (l == 0) {
                if (t < 5) { nA = inputs + (size_t)(t + 1) * 49152; nABS = 393216; }
                else       { nA = A0buf;                            nABS = 49152;  }
            } else { nA = Anext[l]; nABS = (long)chl * hwl; }

            long et = 4L * chl * hwl;
            error_kernel<<<nblk(et), 256, 0, stream>>>(h[l], nA, nABS, Ebuf, chl, hwl, et);
            const int Cin = 3 * chl;
            pack(curA, curBS, cbuf, chl, hwl, Cin, 0);
            pack(Ebuf, 2L * chl * hwl, cbuf, 2 * chl, hwl, Cin, chl);
            conv(cbuf, (long)Cin * hwl, UPD_P[l], UPD_B[l], pooltmp, Cin, CH[l + 1], R);
            long pt = 4L * CH[l + 1] * HW[l + 1];
            maxpool_kernel<<<nblk(pt), 256, 0, stream>>>(pooltmp, Anext[l + 1], CH[l + 1], R, R, pt);
        }

        // ---- loss: mean(cat(relu(P-tgt),relu(tgt-P))^2) == sum((P-tgt)^2)/(N*2C*HW) ----
        {
            const float* tgt = inputs + (size_t)(t + 1) * 49152;
            float lam   = (t == 0) ? 0.5f : 1.0f;
            float scale = lam * 1.0f / 393216.0f;   // WEIGHT_FACTOR=1, N*2*3*128*128
            loss_partial_kernel<<<256, 256, 0, stream>>>(h[0], tgt, 393216L, partial, 4L * 49152, 49152);
            loss_finalize_kernel<<<1, 256, 0, stream>>>(partial, scale, (float*)d_out);
        }

        for (int l = 1; l < 4; ++l) { float* tmp = Acur[l]; Acur[l] = Anext[l]; Anext[l] = tmp; }
    }
}
